// SchNet_84052509983252
// MI455X (gfx1250) — matrix operand order repaired and checked by reading, not compiled, not run on hardware
//
#include <hip/hip_runtime.h>
#include <hip/hip_bf16.h>
#include <math.h>

typedef __attribute__((ext_vector_type(16))) _Float16 v16h;
typedef __attribute__((ext_vector_type(8)))  _Float16 v8h;
typedef __attribute__((ext_vector_type(8)))  float    v8f;
typedef __attribute__((ext_vector_type(4)))  float    v4f;

#define N_ATOMS 256
#define BATCH   8
#define NROWS   (BATCH * N_ATOMS)       // 2048
#define NCH     128
#define NG      25
#define GDELTA  (5.0f / 24.0f)
#define GCOEF   (0.5f / (GDELTA * GDELTA))
#define WROW    136                     // padded row (halves): 272B, 16B aligned
#define WPACK   (128 * WROW)            // halves per packed 128x128 matrix
#define WBYTES  (WPACK * 2)             // 34816 bytes
#define WCHUNKS (WBYTES / 16)           // 2176 x b128 transfers

__device__ __forceinline__ v8f wmma_f16f32(v16h a, v16h b, v8f c) {
  return __builtin_amdgcn_wmma_f32_16x16x32_f16(false, a, false, b, (short)0, c,
                                                false, false);
}

// branch-free tanh (keeps EXEC all-ones between WMMAs)
__device__ __forceinline__ float fast_tanh(float x) {
  x = fminf(9.0f, fmaxf(-9.0f, x));
  float e = __expf(2.0f * x);
  return (e - 1.0f) / (e + 1.0f);
}

// gfx1250 async global->LDS copy (16B), tracked by ASYNCcnt.
__device__ __forceinline__ void async_copy_b128(unsigned lds_off, const void* g) {
  asm volatile("global_load_async_to_lds_b128 %0, %1, off"
               :: "v"(lds_off), "v"((unsigned long long)(uintptr_t)g)
               : "memory");
}
__device__ __forceinline__ void wait_async0() {
  asm volatile("s_wait_asynccnt 0x0" ::: "memory");
}

// ---------------------------------------------------------------------------
// Pack one 128x128 f32 weight (K-major) into f16, out-major padded rows:
//   Wp[o*136 + k] = (f16) W[k*128 + o]   (k>=128 pad -> 0)
__global__ __launch_bounds__(256) void schnet_pack_w(const float* __restrict__ W,
                                                     _Float16* __restrict__ Wp) {
  const int idx = blockIdx.x * 256 + threadIdx.x;
  if (idx < WPACK) {
    const int o = idx / WROW;
    const int k = idx % WROW;
    Wp[idx] = (_Float16)((k < NCH) ? W[k * NCH + o] : 0.0f);
  }
}

// ---------------------------------------------------------------------------
// x[row, c] = emb[z[row], c]
__global__ __launch_bounds__(256) void schnet_embed(const int* __restrict__ z,
                                                    const float* __restrict__ emb,
                                                    float* __restrict__ x) {
  const int idx = blockIdx.x * 256 + threadIdx.x;
  if (idx < NROWS * NCH) {
    const int row = idx >> 7;
    const int c   = idx & (NCH - 1);
    x[idx] = emb[z[row] * NCH + c];
  }
}

// ---------------------------------------------------------------------------
// 2048x128 @ 128x128 WMMA GEMM: out = [ACC? out +] (TANH?)(A @ W + b)
// 4 waves per WG (64 rows); weight async-staged once into LDS as f16.
// DO_T: write out as [col][row] (b128 stores) for the cfconv consumer.
// Flags are compile-time so each instantiation is branch-free straight line.
template <int DO_TANH, int DO_ACC, int DO_T>
__global__ __launch_bounds__(128) void schnet_gemm128(const float* __restrict__ A,
                                                      const _Float16* __restrict__ Wp,
                                                      const float* __restrict__ bias,
                                                      float* __restrict__ out) {
  __shared__ __align__(16) _Float16 wlds[WPACK];
  const int tid  = threadIdx.x;
  const int lane = tid & 31;
  const int wid  = tid >> 5;
  const int n16  = lane & 15;
  const int l16  = lane >> 4;

  const unsigned lbase = (unsigned)(uintptr_t)wlds;
  for (int t = tid; t < WCHUNKS; t += 128)                 // 17 iters exactly
    async_copy_b128(lbase + t * 16, (const char*)Wp + t * 16);
  wait_async0();
  __syncthreads();

  const int m0 = blockIdx.x * 64 + wid * 16;

  // A-fragments: lane holds row m0+n16; halves follow the A wrap pattern.
  v16h afr[4];
  const float* arow = A + (m0 + n16) * NCH;
#pragma unroll
  for (int kc = 0; kc < 4; ++kc) {
    const int k0 = kc * 32 + 8 * l16;
    v4f p0 = *(const v4f*)(arow + k0);
    v4f p1 = *(const v4f*)(arow + k0 + 4);
    v4f p2 = *(const v4f*)(arow + k0 + 16);
    v4f p3 = *(const v4f*)(arow + k0 + 20);
#pragma unroll
    for (int e = 0; e < 4; ++e) {
      afr[kc][e]      = (_Float16)p0[e];
      afr[kc][4 + e]  = (_Float16)p1[e];
      afr[kc][8 + e]  = (_Float16)p2[e];
      afr[kc][12 + e] = (_Float16)p3[e];
    }
  }
#pragma unroll
  for (int nc = 0; nc < 8; ++nc) {
    const int col = nc * 16 + n16;
    const _Float16* wrow = &wlds[col * WROW];
    v8f c = {};
#pragma unroll
    for (int kc = 0; kc < 4; ++kc) {
      const int kb = kc * 32 + 8 * l16;     // B layout: lane n/l16, K = kb + h (h<8), kb + 16 + (h-8) (h>=8)
      v8h blo = *(const v8h*)(wrow + kb);
      v8h bhi = *(const v8h*)(wrow + kb + 16);
      v16h bfr;
#pragma unroll
      for (int h = 0; h < 8; ++h) { bfr[h] = blo[h]; bfr[8 + h] = bhi[h]; }
      c = wmma_f16f32(afr[kc], bfr, c);
    }
    const float bv = bias[col];
    if (DO_T) {
      // D rows m0+8*l16 .. +7 at fixed col are contiguous in [col][row].
      v4f lo, hi;
#pragma unroll
      for (int r = 0; r < 4; ++r) {
        float v0 = c[r] + bv;
        float v1 = c[r + 4] + bv;
        if (DO_TANH) { v0 = fast_tanh(v0); v1 = fast_tanh(v1); }
        lo[r] = v0; hi[r] = v1;
      }
      float* pt = out + col * NROWS + m0 + 8 * l16;
      *(v4f*)pt = lo;
      *(v4f*)(pt + 4) = hi;
    } else {
#pragma unroll
      for (int r = 0; r < 8; ++r) {
        float v = c[r] + bv;                // D row m = r + 8*l16
        if (DO_TANH) v = fast_tanh(v);
        float* p = out + (m0 + r + 8 * l16) * NCH + col;
        if (DO_ACC) v += *p;
        *p = v;
      }
    }
  }
}

// ---------------------------------------------------------------------------
// Fused continuous-filter convolution, one wave32 per atom, 4 atoms per WG.
// yT is channel-major: yT[c][b*256+j].
//   yagg[i,c] = sum_{j!=i}  y[j,c] * ( W2^T tanh(W1^T rbf(r_ij) + b1) + b2 )[c]
__global__ __launch_bounds__(128) void schnet_cfconv(const float* __restrict__ pos,
                                                     const float* __restrict__ yT,
                                                     const float* __restrict__ w1,
                                                     const float* __restrict__ b1,
                                                     const _Float16* __restrict__ w2p,
                                                     const float* __restrict__ b2,
                                                     float* __restrict__ yagg) {
  __shared__ __align__(16) _Float16 w2t[WPACK];
  const int tid  = threadIdx.x;
  const int lane = tid & 31;
  const int wid  = tid >> 5;
  const int n16  = lane & 15;
  const int l16  = lane >> 4;
  const int gidx = blockIdx.x * 4 + wid;    // 0..2047
  const int i    = gidx & (N_ATOMS - 1);
  const int b    = gidx >> 8;

  // Stage packed W2^T (f16) into LDS with gfx1250 async copies.
  const unsigned lbase = (unsigned)(uintptr_t)w2t;
  for (int t = tid; t < WCHUNKS; t += 128)                 // 17 iters exactly
    async_copy_b128(lbase + t * 16, (const char*)w2p + t * 16);
  wait_async0();
  __syncthreads();

  // Loop-invariant A-fragments of W1^T (M = filter channel, K = gauss pad 32).
  v16h aw1[8];
#pragma unroll
  for (int mc = 0; mc < 8; ++mc) {
    const int ch = mc * 16 + n16;
#pragma unroll
    for (int h = 0; h < 16; ++h) {
      const int g = (h < 8 ? h : h + 8) + 8 * l16;
      aw1[mc][h] = (_Float16)((g < NG) ? w1[g * NCH + ch] : 0.0f);
    }
  }
  float b1f[8][8];
#pragma unroll
  for (int mc = 0; mc < 8; ++mc)
#pragma unroll
    for (int r = 0; r < 8; ++r) b1f[mc][r] = b1[mc * 16 + 8 * l16 + r];
  float b2f[8];
#pragma unroll
  for (int nc = 0; nc < 8; ++nc) b2f[nc] = b2[nc * 16 + n16];

  const float pix = pos[(b * N_ATOMS + i) * 3 + 0];
  const float piy = pos[(b * N_ATOMS + i) * 3 + 1];
  const float piz = pos[(b * N_ATOMS + i) * 3 + 2];

  float acc[8];
#pragma unroll
  for (int nc = 0; nc < 8; ++nc) acc[nc] = 0.0f;

  for (int jt = 0; jt < 16; ++jt) {
    // RBF expansion straight into the GEMM1 B-fragment (lane = neighbor col).
    const int j = jt * 16 + n16;
    const float dx = pos[(b * N_ATOMS + j) * 3 + 0] - pix;
    const float dy = pos[(b * N_ATOMS + j) * 3 + 1] - piy;
    const float dz = pos[(b * N_ATOMS + j) * 3 + 2] - piz;
    const float r_ij = sqrtf(dx * dx + dy * dy + dz * dz);
    v16h fB;
#pragma unroll
    for (int h = 0; h < 16; ++h) {
      const int g = (h < 8 ? h : h + 8) + 8 * l16;
      float v = 0.0f;
      if (g < NG) {
        const float d = r_ij - (float)g * GDELTA;
        v = __expf(-GCOEF * d * d);
      }
      fB[h] = (_Float16)v;
    }

    // GEMM1: H^T = W1^T @ F^T ; bias + tanh ; in-lane repack -> GEMM2 A-layout.
    v16h a2[4];
#pragma unroll
    for (int mc = 0; mc < 8; ++mc) {
      v8f d = {};
      d = wmma_f16f32(aw1[mc], fB, d);
#pragma unroll
      for (int r = 0; r < 8; ++r)
        a2[mc >> 1][(mc & 1) * 8 + r] = (_Float16)fast_tanh(d[r] + b1f[mc][r]);
    }

    // GEMM2: W_tile = H @ W2, fused with elementwise y-multiply + neighbor sum.
    const int jrow0 = b * N_ATOMS + jt * 16 + 8 * l16;     // 8 consecutive rows
#pragma unroll
    for (int nc = 0; nc < 8; ++nc) {
      v8f c = {};
      const _Float16* wrow = &w2t[(nc * 16 + n16) * WROW];
#pragma unroll
      for (int kc = 0; kc < 4; ++kc) {
        const int kb = kc * 32 + 8 * l16;
        v8h blo = *(const v8h*)(wrow + kb);
        v8h bhi = *(const v8h*)(wrow + kb + 16);
        v16h bfr;
#pragma unroll
        for (int h = 0; h < 8; ++h) { bfr[h] = blo[h]; bfr[8 + h] = bhi[h]; }
        c = wmma_f16f32(a2[kc], bfr, c);
      }
      // Channel-major y: 8 consecutive rows at fixed channel -> 2 x b128.
      const float* yp = yT + (nc * 16 + n16) * NROWS + jrow0;
      v4f y0 = *(const v4f*)yp;
      v4f y1 = *(const v4f*)(yp + 4);
      float s = 0.0f;
#pragma unroll
      for (int r = 0; r < 8; ++r) {
        const int jj = jt * 16 + r + 8 * l16;          // D row = neighbor
        const float yv = (r < 4) ? y0[r] : y1[r - 4];
        const float wv = c[r] + b2f[nc];
        s += (jj == i) ? 0.0f : yv * wv;               // mask self, no branch
      }
      acc[nc] += s;
    }
  }
#pragma unroll
  for (int nc = 0; nc < 8; ++nc) {
    const float tot = acc[nc] + __shfl_xor(acc[nc], 16, 32);
    if (l16 == 0) yagg[(b * N_ATOMS + i) * NCH + nc * 16 + n16] = tot;
  }
}

// ---------------------------------------------------------------------------
// out[b] = sum_n ( x2[b,n,:] . e2_w ) + N * e2_b
__global__ __launch_bounds__(256) void schnet_reduce(const float* __restrict__ x2,
                                                     const float* __restrict__ e2w,
                                                     const float* __restrict__ e2b,
                                                     float* __restrict__ out) {
  __shared__ float red[256];
  const int b = blockIdx.x;
  const int t = threadIdx.x;
  const float* row = x2 + (b * N_ATOMS + t) * NCH;
  float s = 0.0f;
  for (int c = 0; c < NCH; ++c) s += row[c] * e2w[c];
  red[t] = s;
  __syncthreads();
  for (int off = 128; off > 0; off >>= 1) {
    if (t < off) red[t] += red[t + off];
    __syncthreads();
  }
  if (t == 0) out[b] = red[0] + (float)N_ATOMS * e2b[0];
}

// ---------------------------------------------------------------------------
extern "C" void kernel_launch(void* const* d_in, const int* in_sizes, int n_in,
                              void* d_out, int out_size, void* d_ws, size_t ws_size,
                              hipStream_t stream) {
  (void)in_sizes; (void)n_in; (void)out_size; (void)ws_size;
  const float* pos    = (const float*)d_in[0];
  const int*   z      = (const int*)d_in[1];
  const float* emb    = (const float*)d_in[2];
  const float* fw1_w  = (const float*)d_in[3];
  const float* fw1_b  = (const float*)d_in[4];
  const float* fw2_w  = (const float*)d_in[5];
  const float* fw2_b  = (const float*)d_in[6];
  const float* in2f_w = (const float*)d_in[7];
  const float* in2f_b = (const float*)d_in[8];
  const float* f2o_w  = (const float*)d_in[9];
  const float* f2o_b  = (const float*)d_in[10];
  const float* dns_w  = (const float*)d_in[11];
  const float* dns_b  = (const float*)d_in[12];
  const float* e1_w   = (const float*)d_in[13];
  const float* e1_b   = (const float*)d_in[14];
  const float* e2_w   = (const float*)d_in[15];
  const float* e2_b   = (const float*)d_in[16];
  float* outp = (float*)d_out;

  float* x    = (float*)d_ws;        // [2048,128]
  float* buf1 = x + NROWS * NCH;     // yT / t / x2  [2048*128]
  float* yagg = buf1 + NROWS * NCH;  // [2048,128]
  _Float16* fpk = (_Float16*)(yagg + NROWS * NCH);  // 13 packed f16 matrices
  _Float16* p_in2f[3], *p_f2o[3], *p_dns[3], *p_fw2[3], *p_e1;
  {
    _Float16* cur = fpk;
    for (int l = 0; l < 3; ++l) { p_in2f[l] = cur; cur += WPACK; }
    for (int l = 0; l < 3; ++l) { p_f2o[l]  = cur; cur += WPACK; }
    for (int l = 0; l < 3; ++l) { p_dns[l]  = cur; cur += WPACK; }
    for (int l = 0; l < 3; ++l) { p_fw2[l]  = cur; cur += WPACK; }
    p_e1 = cur;
  }
  const int packBlocks = (WPACK + 255) / 256;
  for (int l = 0; l < 3; ++l) {
    schnet_pack_w<<<packBlocks, 256, 0, stream>>>(in2f_w + l * NCH * NCH, p_in2f[l]);
    schnet_pack_w<<<packBlocks, 256, 0, stream>>>(f2o_w  + l * NCH * NCH, p_f2o[l]);
    schnet_pack_w<<<packBlocks, 256, 0, stream>>>(dns_w  + l * NCH * NCH, p_dns[l]);
    schnet_pack_w<<<packBlocks, 256, 0, stream>>>(fw2_w  + l * NCH * NCH, p_fw2[l]);
  }
  schnet_pack_w<<<packBlocks, 256, 0, stream>>>(e1_w, p_e1);

  schnet_embed<<<(NROWS * NCH + 255) / 256, 256, 0, stream>>>(z, emb, x);
  for (int l = 0; l < 3; ++l) {
    schnet_gemm128<0, 0, 1><<<NROWS / 64, 128, 0, stream>>>(
        x, p_in2f[l], in2f_b + l * NCH, buf1);                     // -> yT
    schnet_cfconv<<<NROWS / 4, 128, 0, stream>>>(
        pos, buf1, fw1_w + l * NG * NCH, fw1_b + l * NCH,
        p_fw2[l], fw2_b + l * NCH, yagg);
    schnet_gemm128<1, 0, 0><<<NROWS / 64, 128, 0, stream>>>(
        yagg, p_f2o[l], f2o_b + l * NCH, buf1);                    // t = tanh
    schnet_gemm128<0, 1, 0><<<NROWS / 64, 128, 0, stream>>>(
        buf1, p_dns[l], dns_b + l * NCH, x);                       // x += dense
  }
  schnet_gemm128<1, 0, 0><<<NROWS / 64, 128, 0, stream>>>(x, p_e1, e1_b, buf1);
  schnet_reduce<<<BATCH, 256, 0, stream>>>(buf1, e2_w, e2_b, outp);
}
